// GeniePathLazy_26121991094922
// MI455X (gfx1250) — matrix operand order repaired and checked by reading, not compiled, not run on hardware
//
#include <hip/hip_runtime.h>
#include <hip/hip_bf16.h>

typedef __attribute__((ext_vector_type(16))) _Float16 v16h;
typedef __attribute__((ext_vector_type(8)))  float    v8f;

#define LDS_STR 72          // halves per staged row (144B: 16B-aligned, conflict-free)
#define CHUNK   64          // K elements staged per pipeline stage

// ---------------------------------------------------------------------------
// CDNA5 async copy helpers (ASYNCcnt-tracked, CDNA5 ISA)
// ---------------------------------------------------------------------------
__device__ __forceinline__ void async_lds_b128(unsigned lds_off, const _Float16* gaddr) {
    asm volatile("global_load_async_to_lds_b128 %0, %1, off"
                 :: "v"(lds_off), "v"(gaddr) : "memory");
}
__device__ __forceinline__ void wait_async0() {
    asm volatile("s_wait_asynccnt 0x0" ::: "memory");
}
__device__ __forceinline__ void wait_async1() {
    asm volatile("s_wait_asynccnt 0x1" ::: "memory");
}

// ---------------------------------------------------------------------------
// utility kernels
// ---------------------------------------------------------------------------
__global__ void fill_u32_kernel(unsigned* __restrict__ p, unsigned v, long long n) {
    long long i = (long long)blockIdx.x * blockDim.x + threadIdx.x;
    if (i < n) p[i] = v;
}

// vectorized f32 -> f16 conversion (n multiple of 4)
__global__ void f32_to_f16_kernel(const float* __restrict__ src,
                                  _Float16* __restrict__ dst, long long n4) {
    long long i = (long long)blockIdx.x * blockDim.x + threadIdx.x;
    if (i >= n4) return;
    float4 v = ((const float4*)src)[i];
    union { _Float16 hh[4]; uint2 u2; } t;
    t.hh[0] = (_Float16)v.x; t.hh[1] = (_Float16)v.y;
    t.hh[2] = (_Float16)v.z; t.hh[3] = (_Float16)v.w;
    ((uint2*)dst)[i] = t.u2;
}

// ---------------------------------------------------------------------------
// WMMA GEMM segment with async double-buffered A staging.
// A: f16 row-major [M x K] (lda == K).  W: f16 [Nc x wstr], cols [wc, wc+K).
// Block 256 thr (8 waves): tile 32(M) x 128(N); wave = 16 cols x 2 M-subtiles.
// ---------------------------------------------------------------------------
__device__ __forceinline__ void gemm_seg(
    v8f* acc, _Float16 (*lds)[32 * LDS_STR],
    const _Float16* __restrict__ A, const _Float16* __restrict__ W, int K,
    int wstr, int wc, int M, int Nc, int m0, int n0w)
{
    const unsigned tid  = threadIdx.x;
    const unsigned lane = tid & 31u;
    const unsigned hf   = lane >> 4;
    const unsigned ln   = lane & 15u;
    const int  ncol     = n0w + (int)ln;
    const bool colValid = (n0w + 16 <= Nc);    // wave-uniform (Nc%16==0)
    const bool fullM    = (m0 + 32 <= M);      // block-uniform
    const int  srow     = (int)tid >> 3;       // 0..31
    const int  scolh    = ((int)tid & 7) * 8;  // 0..56 halves (16B granules)

    const int nch = (K + CHUNK - 1) / CHUNK;

    // stage chunk ch into buffer buf; returns true if staged asynchronously
    auto stage = [&](int ch, int buf) -> bool {
        const int k0 = ch * CHUNK;
        if (fullM && (k0 + CHUNK <= K)) {
            unsigned loff = (unsigned)(size_t)&lds[buf][srow * LDS_STR + scolh];
            async_lds_b128(loff, A + (size_t)(m0 + srow) * K + k0 + scolh);
            return true;
        }
        uint4 v = make_uint4(0u, 0u, 0u, 0u);
        int row = m0 + srow;
        if (row < M && (k0 + scolh + 8) <= K)   // K%16==0 in all uses
            v = *(const uint4*)(A + (size_t)row * K + k0 + scolh);
        *(uint4*)&lds[buf][srow * LDS_STR + scolh] = v;
        return false;
    };

    bool pending = stage(0, 0);
    for (int ic = 0; ic < nch; ++ic) {
        const int buf = ic & 1;
        bool nxt = false;
        if (ic + 1 < nch) nxt = stage(ic + 1, buf ^ 1);
        if (pending) { if (nxt) wait_async1(); else wait_async0(); }
        pending = nxt;
        __syncthreads();   // chunk ic visible to all waves

        const int k0 = ic * CHUNK;
        #pragma unroll
        for (int kh = 0; kh < 2; ++kh) {
            const int kb = k0 + kh * 32;
            if (kb >= K) continue;      // partial chunk (e.g. K == 16)

            // ---- B fragment (32x16): B[k][n] = W[n][wc+kb+k] --------------
            v16h bf;
            unsigned* bfu = reinterpret_cast<unsigned*>(&bf);
            if (colValid && (kb + 32 <= K)) {
                const _Float16* wp = W + (size_t)ncol * wstr + wc + kb + 16 * (int)hf;
                #pragma unroll
                for (int q = 0; q < 8; ++q)
                    bfu[q] = *(const unsigned*)(wp + 2 * q);
            } else {
                #pragma unroll
                for (int q = 0; q < 8; ++q) {
                    int k = 2 * q + 16 * (int)hf;
                    unsigned u = 0u;
                    if (ncol < Nc && (kb + k) < K)
                        u = *(const unsigned*)(W + (size_t)ncol * wstr + wc + kb + k);
                    bfu[q] = u;
                }
            }

            // ---- A fragments + 2 WMMAs -----------------------------------
            #pragma unroll
            for (int sub = 0; sub < 2; ++sub) {
                v16h af;
                unsigned* afu = reinterpret_cast<unsigned*>(&af);
                const int row = (int)ln + 16 * sub;
                #pragma unroll
                for (int q = 0; q < 8; ++q) {
                    int k = 2 * q + 16 * (int)hf + kh * 32;
                    afu[q] = *(const unsigned*)&lds[buf][row * LDS_STR + k];
                }
                acc[sub] = __builtin_amdgcn_wmma_f32_16x16x32_f16(
                    false, af, false, bf, (short)0, acc[sub], false, false);
            }
        }
        __syncthreads();   // readers done before next async overwrites buf
    }
}

// act: 0 = none, 1 = relu, 2 = tanh.  Cf32 / Cf16 optional (either may be null)
__global__ __launch_bounds__(256)
void wmma_gemm3_kernel(
    const _Float16* __restrict__ A0, const _Float16* __restrict__ W0, int K0, int ws0, int wc0,
    const _Float16* __restrict__ A1, const _Float16* __restrict__ W1, int K1, int ws1, int wc1,
    const _Float16* __restrict__ A2, const _Float16* __restrict__ W2, int K2, int ws2, int wc2,
    const float* __restrict__ bias, float* __restrict__ Cf32, _Float16* __restrict__ Cf16,
    int M, int Nc, int act)
{
    __shared__ _Float16 lds_a[2][32 * LDS_STR];

    const unsigned lane = threadIdx.x & 31u;
    const unsigned wv   = threadIdx.x >> 5;
    const unsigned hf   = lane >> 4;
    const unsigned ln   = lane & 15u;
    const int m0   = blockIdx.x * 32;
    const int n0w  = blockIdx.y * 128 + (int)wv * 16;
    const int ncol = n0w + (int)ln;

    v8f acc[2] = {};

    if (A0 != nullptr && K0 > 0) gemm_seg(acc, lds_a, A0, W0, K0, ws0, wc0, M, Nc, m0, n0w);
    if (A1 != nullptr && K1 > 0) gemm_seg(acc, lds_a, A1, W1, K1, ws1, wc1, M, Nc, m0, n0w);
    if (A2 != nullptr && K2 > 0) gemm_seg(acc, lds_a, A2, W2, K2, ws2, wc2, M, Nc, m0, n0w);

    float bv = 0.f;
    if (bias != nullptr && ncol < Nc) bv = bias[ncol];
    #pragma unroll
    for (int sub = 0; sub < 2; ++sub) {
        #pragma unroll
        for (int r = 0; r < 8; ++r) {
            int row = m0 + 16 * sub + r + 8 * (int)hf;
            if (row < M && ncol < Nc) {
                float v = acc[sub][r] + bv;
                if (act == 1)      v = fmaxf(v, 0.f);
                else if (act == 2) v = tanhf(v);
                if (Cf32) Cf32[(size_t)row * Nc + ncol] = v;
                if (Cf16) Cf16[(size_t)row * Nc + ncol] = (_Float16)v;
            }
        }
    }
}

// ---------------------------------------------------------------------------
// AGNN pieces (16-dim f32 features, implicit self-loops at e >= E)
// ---------------------------------------------------------------------------
__device__ __forceinline__ void edge_sd(const long long* ei, long long E,
                                        long long e, int& s, int& d) {
    if (e < E) { s = (int)ei[e]; d = (int)ei[E + e]; }
    else       { s = d = (int)(e - E); }
}

__global__ void node_norm_kernel(const float* __restrict__ h, float* __restrict__ hn, int n) {
    int i = blockIdx.x * blockDim.x + threadIdx.x;
    if (i >= n) return;
    const float4* p = (const float4*)(h + (size_t)i * 16);
    float4 a = p[0], b = p[1], c = p[2], d = p[3];
    float s = a.x*a.x + a.y*a.y + a.z*a.z + a.w*a.w
            + b.x*b.x + b.y*b.y + b.z*b.z + b.w*b.w
            + c.x*c.x + c.y*c.y + c.z*c.z + c.w*c.w
            + d.x*d.x + d.y*d.y + d.z*d.z + d.w*d.w;
    float inv = 1.0f / fmaxf(sqrtf(s), 1e-12f);
    float4* o = (float4*)(hn + (size_t)i * 16);
    o[0] = make_float4(a.x*inv, a.y*inv, a.z*inv, a.w*inv);
    o[1] = make_float4(b.x*inv, b.y*inv, b.z*inv, b.w*inv);
    o[2] = make_float4(c.x*inv, c.y*inv, c.z*inv, c.w*inv);
    o[3] = make_float4(d.x*inv, d.y*inv, d.z*inv, d.w*inv);
}

__global__ void edge_logit_kernel(const long long* __restrict__ ei, long long E, long long EE,
                                  const float* __restrict__ hn, const float* __restrict__ beta_ptr,
                                  float* __restrict__ logit, unsigned* __restrict__ segmax) {
    long long e = (long long)blockIdx.x * blockDim.x + threadIdx.x;
    if (e >= EE) return;
    int s, d; edge_sd(ei, E, e, s, d);
    const float4* ps = (const float4*)(hn + (size_t)s * 16);
    const float4* pd = (const float4*)(hn + (size_t)d * 16);
    float dot = 0.f;
    #pragma unroll
    for (int q = 0; q < 4; ++q) {
        float4 a = ps[q], b = pd[q];
        dot += a.x*b.x + a.y*b.y + a.z*b.z + a.w*b.w;
    }
    float beta = beta_ptr ? beta_ptr[0] : 1.0f;
    float l = beta * dot;
    logit[e] = l;
    int bi = __float_as_int(l);
    unsigned key = (bi >= 0) ? ((unsigned)bi | 0x80000000u) : ~(unsigned)bi;
    atomicMax(&segmax[d], key);
}

__global__ void edge_expsum_kernel(const long long* __restrict__ ei, long long E, long long EE,
                                   float* __restrict__ logit, const unsigned* __restrict__ segmax,
                                   float* __restrict__ segsum) {
    long long e = (long long)blockIdx.x * blockDim.x + threadIdx.x;
    if (e >= EE) return;
    int s, d; edge_sd(ei, E, e, s, d);
    unsigned k = segmax[d];
    float m = (k & 0x80000000u) ? __int_as_float((int)(k & 0x7fffffffu))
                                : __int_as_float((int)~k);
    float ex = expf(logit[e] - m);
    logit[e] = ex;
    atomicAdd(&segsum[d], ex);
}

__global__ void edge_agg_kernel(const long long* __restrict__ ei, long long E, long long EE,
                                const float* __restrict__ ex, const float* __restrict__ segsum,
                                const float* __restrict__ hin, float* __restrict__ out) {
    long long t = (long long)blockIdx.x * blockDim.x + threadIdx.x;
    if (t >= EE * 16) return;
    long long e = t >> 4;
    int dim = (int)(t & 15);
    int s, d; edge_sd(ei, E, e, s, d);
    float alpha = ex[e] / fmaxf(segsum[d], 1e-16f);
    atomicAdd(&out[(size_t)d * 16 + dim], alpha * hin[(size_t)s * 16 + dim]);
}

// ---------------------------------------------------------------------------
// LSTM pointwise (gate order i,f,g,o; H = 256). h state kept in f16 (only
// ever consumed as a GEMM A operand); c state in f32.
// ---------------------------------------------------------------------------
__global__ void lstm_point_kernel(const float* __restrict__ g, _Float16* __restrict__ h16,
                                  float* __restrict__ c, int n) {
    int idx = blockIdx.x * blockDim.x + threadIdx.x;
    if (idx >= n) return;
    int node = idx >> 8;
    int j = idx & 255;
    const float* gr = g + (size_t)node * 1024;
    float ig = gr[j], fg = gr[256 + j], gg = gr[512 + j], og = gr[768 + j];
    float si = 1.f / (1.f + expf(-ig));
    float sf = 1.f / (1.f + expf(-fg));
    float so = 1.f / (1.f + expf(-og));
    float cv = sf * c[idx] + si * tanhf(gg);
    c[idx] = cv;
    h16[idx] = (_Float16)(so * tanhf(cv));
}

// ---------------------------------------------------------------------------
// host helpers
// ---------------------------------------------------------------------------
static void run_agnn(const long long* ei, long long E, long long EE, int N,
                     const float* in16, float* out16, float* hn, float* segsum,
                     unsigned* segmax, float* logit, const float* beta_ptr,
                     hipStream_t stream) {
    dim3 blk(256);
    unsigned g16 = (unsigned)(((long long)N * 16 + 255) / 256);
    unsigned g2  = (unsigned)((2LL * N + 255) / 256);
    unsigned gN  = (unsigned)((N + 255) / 256);
    unsigned gE  = (unsigned)((EE + 255) / 256);
    unsigned gA  = (unsigned)((EE * 16 + 255) / 256);
    fill_u32_kernel<<<g16, blk, 0, stream>>>((unsigned*)out16, 0u, (long long)N * 16);
    fill_u32_kernel<<<g2,  blk, 0, stream>>>((unsigned*)segsum, 0u, 2LL * N); // segsum+segmax
    node_norm_kernel<<<gN, blk, 0, stream>>>(in16, hn, N);
    edge_logit_kernel<<<gE, blk, 0, stream>>>(ei, E, EE, hn, beta_ptr, logit, segmax);
    edge_expsum_kernel<<<gE, blk, 0, stream>>>(ei, E, EE, logit, segmax, segsum);
    edge_agg_kernel<<<gA, blk, 0, stream>>>(ei, E, EE, logit, segsum, in16, out16);
}

static void conv16(const float* src, _Float16* dst, long long n, hipStream_t stream) {
    long long n4 = n / 4;
    unsigned g = (unsigned)((n4 + 255) / 256);
    f32_to_f16_kernel<<<g, dim3(256), 0, stream>>>(src, dst, n4);
}

// ---------------------------------------------------------------------------
extern "C" void kernel_launch(void* const* d_in, const int* in_sizes, int n_in,
                              void* d_out, int out_size, void* d_ws, size_t ws_size,
                              hipStream_t stream) {
    const float*     x      = (const float*)d_in[0];
    const long long* ei     = (const long long*)d_in[1];   // int64 per reference
    const float*     lin1_W = (const float*)d_in[2];
    const float*     lin1_b = (const float*)d_in[3];
    const float*     bW1    = (const float*)d_in[4];
    const float*     bb1    = (const float*)d_in[5];
    const float*     beta2  = (const float*)d_in[6];
    const float*     bW2    = (const float*)d_in[7];
    const float*     bb2    = (const float*)d_in[8];
    const float*     Wih    = (const float*)d_in[9];
    const float*     Whh    = (const float*)d_in[10];
    const float*     lin2_W = (const float*)d_in[11];
    const float*     lin2_b = (const float*)d_in[12];

    const int IN = 256, D = 256, H = 256, L = 4, BR = 16, OUTD = 128;
    const int N = in_sizes[0] / IN;
    const long long E  = (long long)in_sizes[1] / 2;
    const long long EE = E + N;
    const long long EEpad = (EE + 3) & ~3LL;   // keep f16 region 16B aligned

    // ---------------- workspace layout ----------------
    float* ws   = (float*)d_ws;
    size_t nd   = (size_t)N * D;
    float* c     = ws;                              // N*H          f32
    float* gates = ws + nd;                         // N*4H         f32
    float* h16a  = ws + 5 * nd;                     // N*16         f32 (agnn)
    float* hn    = h16a + (size_t)N * BR;           // N*16
    float* h16b  = hn   + (size_t)N * BR;           // N*16
    float* segsum = h16b + (size_t)N * BR;          // N
    unsigned* segmax = (unsigned*)(segsum + N);     // N
    float* logit  = (float*)(segmax + N);           // EEpad

    // f16 region (16B aligned)
    _Float16* f16r    = (_Float16*)(logit + EEpad);
    _Float16* x16     = f16r;                                  // N*IN
    _Float16* x0_16   = x16    + nd;                           // N*D
    _Float16* htmp16  = x0_16  + nd;                           // N*D
    _Float16* h_16    = htmp16 + nd;                           // N*H (LSTM h)
    _Float16* hbr16   = h_16   + nd;                           // N*16 (breadth agg)
    _Float16* w_lin1  = hbr16  + (size_t)N * BR;               // D*IN
    _Float16* w_b1    = w_lin1 + (size_t)D * IN;               // L*BR*D
    _Float16* w_b2    = w_b1   + (size_t)L * BR * D;           // L*D*BR
    _Float16* w_ih    = w_b2   + (size_t)L * D * BR;           // L*4H*2D
    _Float16* w_hh    = w_ih   + (size_t)L * 4 * H * 2 * D;    // L*4H*H
    _Float16* w_lin2  = w_hh   + (size_t)L * 4 * H * H;        // OUTD*D

    dim3 blk(256);
    const unsigned gm = (unsigned)((N + 31) / 32);

    // ---------------- f16 conversions (weights + input x) ------------------
    conv16(x,      x16,    (long long)N * IN, stream);
    conv16(lin1_W, w_lin1, (long long)D * IN, stream);
    conv16(bW1,    w_b1,   (long long)L * BR * D, stream);
    conv16(bW2,    w_b2,   (long long)L * D * BR, stream);
    conv16(Wih,    w_ih,   (long long)L * 4 * H * 2 * D, stream);
    conv16(Whh,    w_hh,   (long long)L * 4 * H * H, stream);
    conv16(lin2_W, w_lin2, (long long)OUTD * D, stream);

    // zero c (f32) and h_16 (f16 zeros == 0x0000)
    {
        unsigned g = (unsigned)(((long long)nd + 255) / 256);
        fill_u32_kernel<<<g, blk, 0, stream>>>((unsigned*)c, 0u, (long long)nd);
        unsigned g2 = (unsigned)(((long long)nd / 2 + 255) / 256);
        fill_u32_kernel<<<g2, blk, 0, stream>>>((unsigned*)h_16, 0u, (long long)nd / 2);
    }

    // x0 = x @ lin1_W^T + lin1_b   (f16 out only; consumed solely as GEMM A)
    {
        dim3 grid(gm, (unsigned)((D + 127) / 128));
        wmma_gemm3_kernel<<<grid, blk, 0, stream>>>(
            x16, w_lin1, IN, IN, 0,
            nullptr, nullptr, 0, 0, 0,
            nullptr, nullptr, 0, 0, 0,
            lin1_b, nullptr, x0_16, N, D, 0);
    }

    for (int i = 0; i < L; ++i) {
        const _Float16* W1 = w_b1 + (size_t)i * BR * D;
        const float*    b1 = bb1  + (size_t)i * BR;
        const _Float16* W2 = w_b2 + (size_t)i * D * BR;
        const float*    b2 = bb2  + (size_t)i * D;
        const _Float16* Wi = w_ih + (size_t)i * (4 * H) * (2 * D);
        const _Float16* Wh = w_hh + (size_t)i * (4 * H) * H;

        // h16a = relu(x0 @ W1^T + b1)   [N,16] f32 for the edge kernels
        {
            dim3 grid(gm, 1);
            wmma_gemm3_kernel<<<grid, blk, 0, stream>>>(
                x0_16, W1, D, D, 0,
                nullptr, nullptr, 0, 0, 0,
                nullptr, nullptr, 0, 0, 0,
                b1, h16a, nullptr, N, BR, 1);
        }
        // two AGNN propagations (prop1 beta=1, prop2 beta=beta2[i])
        run_agnn(ei, E, EE, N, h16a, h16b, hn, segsum, segmax, logit, nullptr,   stream);
        run_agnn(ei, E, EE, N, h16b, h16a, hn, segsum, segmax, logit, beta2 + i, stream);
        // f16 shadow of aggregated features for the K=16 GEMM
        conv16(h16a, hbr16, (long long)N * BR, stream);

        // htmp = tanh(hbr @ W2^T + b2)   [N,256] f16 (GEMM-A only)
        {
            dim3 grid(gm, (unsigned)((D + 127) / 128));
            wmma_gemm3_kernel<<<grid, blk, 0, stream>>>(
                hbr16, W2, BR, BR, 0,
                nullptr, nullptr, 0, 0, 0,
                nullptr, nullptr, 0, 0, 0,
                b2, nullptr, htmp16, N, D, 2);
        }

        // gates = [htmp, xc] @ Wih^T + h @ Whh^T      [N,1024] f32
        const _Float16* xc = (i == 0) ? x0_16 : h_16;
        {
            dim3 grid(gm, (unsigned)((4 * H + 127) / 128));
            wmma_gemm3_kernel<<<grid, blk, 0, stream>>>(
                htmp16, Wi, D, 2 * D, 0,
                xc,     Wi, D, 2 * D, D,
                h_16,   Wh, H, H,     0,
                nullptr, gates, nullptr, N, 4 * H, 0);
        }
        // LSTM pointwise update of h (f16) and c (f32)
        {
            unsigned g = (unsigned)(((long long)N * H + 255) / 256);
            lstm_point_kernel<<<g, blk, 0, stream>>>(gates, h_16, c, N * H);
        }
    }

    // out = h @ lin2_W^T + lin2_b            [N,128] f32
    {
        dim3 grid(gm, (unsigned)((OUTD + 127) / 128));
        wmma_gemm3_kernel<<<grid, blk, 0, stream>>>(
            h_16, w_lin2, D, D, 0,
            nullptr, nullptr, 0, 0, 0,
            nullptr, nullptr, 0, 0, 0,
            lin2_b, (float*)d_out, nullptr, N, OUTD, 0);
    }

    (void)n_in; (void)out_size; (void)ws_size;
}
